// style_bottleneck_IR_67637144977440
// MI455X (gfx1250) — compile-verified, matrix operand order checked by reading, not run on hardware
//
#include <hip/hip_runtime.h>
#include <hip/hip_bf16.h>

// ---------------------------------------------------------------------------
// StyleGAN-ish bottleneck on MI455X (gfx1250): all convs as implicit GEMM via
// v_wmma_f32_16x16x32_f16 (f16 operands, f32 accumulate).
//   out[M=Cout][N=B*Ho*Wo] = W[M][K=Cin*KS*KS] x im2col[K][N]
// Block: 256 thr (8 waves). Tile: 128(M) x 64(N). Each wave: 16x64 -> 4 WMMA
// per 32-wide K-step. Double-buffered LDS, one barrier per K-step. All dims
// are compile-time powers of two -> no runtime division sequences.
// ---------------------------------------------------------------------------

typedef __attribute__((ext_vector_type(16))) _Float16 v16h;
typedef __attribute__((ext_vector_type(8)))  _Float16 h8;
typedef __attribute__((ext_vector_type(8)))  float    v8f;

#define POST_NONE  0
#define POST_PRELU 1
#define POST_RES   2
#define POST_MOD   3

// Assemble a 16-wide f16 fragment from LDS per the CDNA5 16-bit A/B layout:
// elements [0:8] = K {kb..kb+7}, elements [8:16] = K {kb+16..kb+23}.
__device__ __forceinline__ v16h frag_from_lds(const _Float16* p) {
    h8 lo = *(const h8*)(p);
    h8 hi = *(const h8*)(p + 16);
    v16h r;
#pragma unroll
    for (int i = 0; i < 8; ++i) { r[i] = lo[i]; r[i + 8] = hi[i]; }
    return r;
}

template <int KS, int STRIDE, int PAD, int POST,
          int CIN, int COUT, int H, int W, int HO, int WO>
__global__ __launch_bounds__(256)
void conv_wmma(const float* __restrict__ in, const _Float16* __restrict__ wgt,
               float* __restrict__ out,
               const float* __restrict__ aux,       // prelu alpha / residual / demod(b,co)
               const float* __restrict__ bias,      // act_bias (POST_MOD)
               const float* __restrict__ in_scale)  // s[b][cin]  (POST_MOD)
{
    constexpr int KK    = KS * KS;
    constexpr int Ktot  = CIN * KK;
    constexpr int HoWo  = HO * WO;           // power of two -> shifts
    constexpr int NSTEP = Ktot / 32;

    __shared__ _Float16 ash[2][128][40];     // weight tiles  (rows 80B, 16B-mult)
    __shared__ _Float16 bsh[2][64][40];      // im2col tiles, transposed [n][k]

    const int t    = threadIdx.x;
    const int lane = t & 31;
    const int wv   = t >> 5;
    const int m_base = blockIdx.y * 128;
    const int n_base = blockIdx.x * 64;

    // ---- B-tile staging mapping (n fixed per thread; all divides constant) ----
    const int nn_b  = t & 63;
    const int kk0_b = t >> 6;                // + 4*j below
    const int n_g   = n_base + nn_b;
    const int b_b   = n_g / HoWo;
    const int hwr   = n_g % HoWo;
    const int ho    = hwr / WO;
    const int wo    = hwr % WO;
    const int hi0   = ho * STRIDE - PAD;
    const int wi0   = wo * STRIDE - PAD;
    const float* in_b  = in + (size_t)b_b * CIN * H * W;
    const float* scl_b = (POST == POST_MOD) ? (in_scale + (size_t)b_b * CIN) : nullptr;

    // ---- A-tile staging mapping: 16B f16 chunks ----
    const int kq8 = (t & 3) * 8;             // f16 offset within 32-wide K chunk
    const int ra  = t >> 2;                  // rows 0..63, and +64

    auto stage = [&](int k0, int p) {
        // weights: contiguous [COUT][Ktot] f16, two b128 loads + b128 LDS stores
        const _Float16* wp = wgt + (size_t)(m_base + ra) * Ktot + k0 + kq8;
        *(h8*)&ash[p][ra][kq8]      = *(const h8*)wp;
        *(h8*)&ash[p][ra + 64][kq8] = *(const h8*)(wp + (size_t)64 * Ktot);
        // im2col: 8 elements per thread (groups of 64 threads share kk -> coalesced)
#pragma unroll
        for (int j = 0; j < 8; ++j) {
            const int kk  = kk0_b + 4 * j;
            const int k   = k0 + kk;
            const int cin = k / KK;          // constant divisor (9 or 1)
            const int rr  = k - cin * KK;
            const int kh  = rr / KS;
            const int kw  = rr - kh * KS;
            const int hi  = hi0 + kh;
            const int wi  = wi0 + kw;
            float v = 0.f;
            if ((unsigned)hi < (unsigned)H && (unsigned)wi < (unsigned)W)
                v = in_b[((size_t)cin * H + hi) * W + wi];
            if (POST == POST_MOD)            // modulated conv: scale inputs by s[b][ci]
                v *= scl_b[cin];
            bsh[p][nn_b][kk] = (_Float16)v;
        }
    };

    v8f acc[4];
#pragma unroll
    for (int q = 0; q < 4; ++q)
#pragma unroll
        for (int i = 0; i < 8; ++i) acc[q][i] = 0.f;

    stage(0, 0);
    __syncthreads();
    int p = 0;
    for (int s = 0; s < NSTEP; ++s) {
        if (s + 1 < NSTEP) stage((s + 1) * 32, p ^ 1);  // fill other buffer

        const int rowA = wv * 16 + (lane & 15);
        const int kb   = (lane >> 4) * 8;
        const v16h a = frag_from_lds(&ash[p][rowA][kb]);
#pragma unroll
        for (int q = 0; q < 4; ++q) {
            const v16h b = frag_from_lds(&bsh[p][q * 16 + (lane & 15)][kb]);
            acc[q] = __builtin_amdgcn_wmma_f32_16x16x32_f16(false, a, false, b,
                                                            (short)0, acc[q], false, false);
        }
        __syncthreads();
        p ^= 1;
    }

    // ---- store C fragments with fused post-ops ----
    const int col   = lane & 15;
    const int mrow0 = m_base + wv * 16 + (lane >> 4) * 8;
#pragma unroll
    for (int q = 0; q < 4; ++q) {
        const int n  = n_base + q * 16 + col;
        const int bb = n / HoWo;
        const int h2 = n % HoWo;
        const int oh = h2 / WO;
        const int ow = h2 % WO;
#pragma unroll
        for (int i = 0; i < 8; ++i) {
            float v = acc[q][i];
            const int m = mrow0 + i;
            const size_t oidx = (((size_t)bb * COUT + m) * HO + oh) * WO + ow;
            if (POST == POST_PRELU) {
                const float al = aux[m];
                v = (v >= 0.f) ? v : al * v;
            } else if (POST == POST_RES) {
                v += aux[oidx];
            } else if (POST == POST_MOD) {
                v = v * aux[(size_t)bb * COUT + m] + bias[m];  // demod*w_scale, +act_bias
                v = ((v >= 0.f) ? v : 0.2f * v) * 1.4142135623730951f;
            }
            out[oidx] = v;
        }
    }
}

// ---------------------------------------------------------------------------
// Small helper kernels (negligible FLOPs)
// ---------------------------------------------------------------------------

__global__ void f32_to_f16_kernel(const float* __restrict__ a, _Float16* __restrict__ o, int n)
{
    const int i = blockIdx.x * blockDim.x + threadIdx.x;
    if (i < n) o[i] = (_Float16)a[i];
}

__global__ void downsample2x(const float* __restrict__ x, float* __restrict__ o,
                             int total, int C, int Hs, int Ws, int Ho, int Wo)
{
    const int i = blockIdx.x * blockDim.x + threadIdx.x;
    if (i >= total) return;
    const int w = i % Wo;
    const int h = (i / Wo) % Ho;
    const int c = (i / (Wo * Ho)) % C;
    const int b = i / (Wo * Ho * C);
    o[i] = x[(((size_t)b * C + c) * Hs + 2 * h) * Ws + 2 * w];
}

// s[b][d] = dot(style[b], mod_w[d]) / sqrt(S) + mod_b[d]
__global__ void style_kernel(const float* __restrict__ style, const float* __restrict__ mw,
                             const float* __restrict__ mb, float* __restrict__ s,
                             int B, int D, int S)
{
    const int i = blockIdx.x * blockDim.x + threadIdx.x;
    if (i >= B * D) return;
    const int b = i / D, d = i - b * D;
    const float* sr = style + (size_t)b * S;
    const float* wr = mw + (size_t)d * S;
    float acc = 0.f;
    for (int k = 0; k < S; ++k) acc += sr[k] * wr[k];
    s[i] = acc * (1.0f / sqrtf((float)S)) + mb[d];
}

// q[co][ci] = sum_{kh,kw} w_mc^2
__global__ void qsum_kernel(const float* __restrict__ wmc, float* __restrict__ q, int total)
{
    const int i = blockIdx.x * blockDim.x + threadIdx.x;
    if (i >= total) return;
    const float* p = wmc + (size_t)i * 9;
    float acc = 0.f;
#pragma unroll
    for (int j = 0; j < 9; ++j) acc += p[j] * p[j];
    q[i] = acc;
}

// demod'[b][co] = w_scale * rsqrt(w_scale^2 * sum_ci s^2 * q + 1e-8)
__global__ void demod_kernel(const float* __restrict__ s, const float* __restrict__ q,
                             float* __restrict__ dm, int B, int D, int Cin)
{
    const int i = blockIdx.x * blockDim.x + threadIdx.x;
    if (i >= B * D) return;
    const int b = i / D, co = i - b * D;
    const float* sr = s + (size_t)b * Cin;
    const float* qr = q + (size_t)co * Cin;
    float acc = 0.f;
    for (int k = 0; k < Cin; ++k) { const float sv = sr[k]; acc += sv * sv * qr[k]; }
    const float wsc = 1.0f / sqrtf((float)(Cin * 9));
    dm[i] = wsc / sqrtf(wsc * wsc * acc + 1e-8f);
}

// ---------------------------------------------------------------------------

extern "C" void kernel_launch(void* const* d_in, const int* in_sizes, int n_in,
                              void* d_out, int out_size, void* d_ws, size_t ws_size,
                              hipStream_t stream)
{
    (void)in_sizes; (void)n_in; (void)out_size; (void)ws_size;

    const float* x     = (const float*)d_in[0];   // (16,256,64,64)
    const float* style = (const float*)d_in[1];   // (16,512)
    const float* w1a   = (const float*)d_in[2];   // (256,256,3,3)
    const float* p1    = (const float*)d_in[3];   // (256,)
    const float* w1b   = (const float*)d_in[4];   // (256,256,3,3)
    const float* w2a   = (const float*)d_in[5];   // (512,256,3,3)
    const float* p2    = (const float*)d_in[6];   // (512,)
    const float* w2b   = (const float*)d_in[7];   // (512,512,3,3)
    const float* ws2   = (const float*)d_in[8];   // (512,256,1,1)
    const float* mod_w = (const float*)d_in[9];   // (512,512)
    const float* mod_b = (const float*)d_in[10];  // (512,)
    const float* w_mc  = (const float*)d_in[11];  // (512,512,3,3)
    const float* abias = (const float*)d_in[12];  // (512,)
    float* out = (float*)d_out;                   // (16,512,16,16) fp32

    constexpr int Bn = 16, C1 = 256, C2 = 512;

    // ---- workspace layout: fp32 region then f16 weight region ----
    float* ws   = (float*)d_ws;
    float* h1   = ws; ws += (size_t)Bn * C1 * 64 * 64;
    float* sc   = ws; ws += (size_t)Bn * C1 * 32 * 32;
    float* x1   = ws; ws += (size_t)Bn * C1 * 32 * 32;
    float* h2   = ws; ws += (size_t)Bn * C2 * 32 * 32;
    float* sc2  = ws; ws += (size_t)Bn * C2 * 16 * 16;
    float* x2   = ws; ws += (size_t)Bn * C2 * 16 * 16;
    float* sbuf = ws; ws += (size_t)Bn * C2;
    float* qbuf = ws; ws += (size_t)C2 * C2;
    float* dmod = ws; ws += (size_t)Bn * C2;

    _Float16* hw   = (_Float16*)ws;
    _Float16* w1ah = hw; hw += (size_t)C1 * C1 * 9;
    _Float16* w1bh = hw; hw += (size_t)C1 * C1 * 9;
    _Float16* w2ah = hw; hw += (size_t)C2 * C1 * 9;
    _Float16* w2bh = hw; hw += (size_t)C2 * C2 * 9;
    _Float16* ws2h = hw; hw += (size_t)C2 * C1;
    _Float16* wmch = hw; hw += (size_t)C2 * C2 * 9;

    const dim3 blk(256);
    auto cvt = [&](const float* src, _Float16* dst, int n) {
        f32_to_f16_kernel<<<(n + 255) / 256, blk, 0, stream>>>(src, dst, n);
    };
    cvt(w1a, w1ah, C1 * C1 * 9);
    cvt(w1b, w1bh, C1 * C1 * 9);
    cvt(w2a, w2ah, C2 * C1 * 9);
    cvt(w2b, w2bh, C2 * C2 * 9);
    cvt(ws2, ws2h, C2 * C1);
    cvt(w_mc, wmch, C2 * C2 * 9);

    // sc = x[:, :, ::2, ::2]
    {
        const int total = Bn * C1 * 32 * 32;
        downsample2x<<<(total + 255) / 256, blk, 0, stream>>>(x, sc, total, C1, 64, 64, 32, 32);
    }

    // h1 = prelu(conv3x3_s1(x, w1a), p1)                       [16,256,64,64]
    conv_wmma<3, 1, 1, POST_PRELU, 256, 256, 64, 64, 64, 64>
        <<<dim3(Bn * 64 * 64 / 64, 256 / 128), blk, 0, stream>>>(
            x, w1ah, h1, p1, nullptr, nullptr);

    // x1 = conv3x3_s2(h1, w1b) + sc                            [16,256,32,32]
    conv_wmma<3, 2, 1, POST_RES, 256, 256, 64, 64, 32, 32>
        <<<dim3(Bn * 32 * 32 / 64, 256 / 128), blk, 0, stream>>>(
            h1, w1bh, x1, sc, nullptr, nullptr);

    // sc2 = conv1x1_s2(x1, ws2)                                [16,512,16,16]
    conv_wmma<1, 2, 0, POST_NONE, 256, 512, 32, 32, 16, 16>
        <<<dim3(Bn * 16 * 16 / 64, 512 / 128), blk, 0, stream>>>(
            x1, ws2h, sc2, nullptr, nullptr, nullptr);

    // h2 = prelu(conv3x3_s1(x1, w2a), p2)                      [16,512,32,32]
    conv_wmma<3, 1, 1, POST_PRELU, 256, 512, 32, 32, 32, 32>
        <<<dim3(Bn * 32 * 32 / 64, 512 / 128), blk, 0, stream>>>(
            x1, w2ah, h2, p2, nullptr, nullptr);

    // x2 = conv3x3_s2(h2, w2b) + sc2                           [16,512,16,16]
    conv_wmma<3, 2, 1, POST_RES, 512, 512, 32, 32, 16, 16>
        <<<dim3(Bn * 16 * 16 / 64, 512 / 128), blk, 0, stream>>>(
            h2, w2bh, x2, sc2, nullptr, nullptr);

    // Style modulation path (tiny)
    style_kernel<<<(Bn * C2 + 255) / 256, blk, 0, stream>>>(style, mod_w, mod_b, sbuf, Bn, C2, C2);
    qsum_kernel<<<(C2 * C2 + 255) / 256, blk, 0, stream>>>(w_mc, qbuf, C2 * C2);
    demod_kernel<<<(Bn * C2 + 255) / 256, blk, 0, stream>>>(sbuf, qbuf, dmod, Bn, C2, C2);

    // out = lrelu( demod * w_scale * conv3x3(x2 * s, w_mc) + bias ) * sqrt(2)
    conv_wmma<3, 1, 1, POST_MOD, 512, 512, 16, 16, 16, 16>
        <<<dim3(Bn * 16 * 16 / 64, 512 / 128), blk, 0, stream>>>(
            x2, wmch, out, dmod, abias, sbuf);
}
